// CrossAttention_13443247636609
// MI455X (gfx1250) — matrix operand order repaired and checked by reading, not compiled, not run on hardware
//
#include <hip/hip_runtime.h>

// CDNA5 / gfx1250 cross-attention, bf16 WMMA everywhere, f32 accumulate.
// Shapes fixed by the reference: B=4, T=S=2048, DIM=1024, CTX=768, H=16, Hd=64.
// Round 2: software-pipelined loads so WMMAs don't sit behind s_wait_loadcnt 0.

#define BSZ   4
#define TSEQ  2048
#define DIMN  1024
#define CTXD  768
#define NH    16
#define HD    64

typedef __attribute__((ext_vector_type(4)))  unsigned int u32x4;
typedef __attribute__((ext_vector_type(16))) __bf16       bf16x16;
typedef __attribute__((ext_vector_type(8)))  float        v8f;

__device__ __forceinline__ unsigned short f2bf(float f) {
  unsigned int u = __builtin_bit_cast(unsigned int, f);
  u += 0x7fffu + ((u >> 16) & 1u);          // round-to-nearest-even
  return (unsigned short)(u >> 16);
}

union BfPack { u32x4 q[2]; bf16x16 v; };

// Build a 16x(bf16) WMMA operand from two 16-byte chunks (global or LDS).
__device__ __forceinline__ bf16x16 load2(const unsigned short* p0,
                                         const unsigned short* p1) {
  BfPack pk;
  pk.q[0] = *reinterpret_cast<const u32x4*>(p0);
  pk.q[1] = *reinterpret_cast<const u32x4*>(p1);
  return pk.v;
}

__device__ __forceinline__ v8f wmma_bf16(bf16x16 a, bf16x16 b, v8f c) {
  // v_wmma_f32_16x16x32_bf16 : D = A(16x32) x B(32x16) + C
  return __builtin_amdgcn_wmma_f32_16x16x32_bf16(
      false, a, false, b, (short)0, c, false, false);
}

// ---------------------------------------------------------------------------
// Elementwise f32 -> bf16
__global__ __launch_bounds__(256) void cvt_bf16(const float* __restrict__ in,
                                                unsigned short* __restrict__ out,
                                                int n) {
  int i = blockIdx.x * 256 + threadIdx.x;
  if (i < n) out[i] = f2bf(in[i]);
}

// Transpose-convert weight W(K,1024) f32 -> Wt(1024,K) bf16  (coalesced writes)
__global__ __launch_bounds__(256) void cvt_tr(const float* __restrict__ W,
                                              unsigned short* __restrict__ Wt,
                                              int K) {
  int i = blockIdx.x * 256 + threadIdx.x;   // i = n*K + k
  int n = i / K, k = i - n * K;
  if (n < DIMN) Wt[i] = f2bf(W[(size_t)k * DIMN + n]);
}

// ---------------------------------------------------------------------------
// GEMM: C(M,1024) = A(M,K) @ Bt(1024,K)^T + bias, bf16 inputs, f32 accumulate.
// One wave -> 16x64 output tile (4 WMMA accumulators). Block = 8 waves.
// Double-buffered K loop: load step kb+32 while multiplying step kb.
// MODE 0: f32 row-major out      (final projection)
// MODE 1: bf16 scatter (b,h,t,d) (Q / K projections)
// MODE 2: bf16 scatter (b,h,d,s) (V projection, transposed for P*V operand)
template <int MODE>
__global__ __launch_bounds__(256, 1) void gemm_bf16(
    const unsigned short* __restrict__ A, const unsigned short* __restrict__ Bt,
    const float* __restrict__ bias, unsigned short* __restrict__ outb,
    float* __restrict__ outf, int M, int K) {
  const int lane  = threadIdx.x & 31;
  const int wv    = threadIdx.x >> 5;
  const int r     = lane & 15;      // row-in-tile / col-in-tile
  const int hh    = lane >> 4;      // K-half select (operand layout)
  const int mtile = blockIdx.x * 16;
  const int nbase = blockIdx.y * 512 + wv * 64;

  const size_t arow = (size_t)(mtile + r) * K;
  size_t brow[4];
  float  bv[4];
#pragma unroll
  for (int j = 0; j < 4; ++j) {
    int n   = nbase + j * 16 + r;
    brow[j] = (size_t)n * K;
    bv[j]   = bias[n];
  }

  v8f acc[4] = {v8f{}, v8f{}, v8f{}, v8f{}};

  // prologue: stage k-step 0
  bf16x16 a_cur = load2(A + arow + hh * 16, A + arow + hh * 16 + 8);
  bf16x16 b_cur[4];
#pragma unroll
  for (int j = 0; j < 4; ++j)
    b_cur[j] = load2(Bt + brow[j] + hh * 16, Bt + brow[j] + hh * 16 + 8);

  for (int kb = 0; kb < K - 32; kb += 32) {
    const int kn = kb + 32;
    // stage next k-step into fresh registers (hides load latency behind WMMA)
    bf16x16 a_nxt = load2(A + arow + kn + hh * 16, A + arow + kn + hh * 16 + 8);
    bf16x16 b_nxt[4];
#pragma unroll
    for (int j = 0; j < 4; ++j)
      b_nxt[j] = load2(Bt + brow[j] + kn + hh * 16,
                       Bt + brow[j] + kn + hh * 16 + 8);
#pragma unroll
    for (int j = 0; j < 4; ++j) acc[j] = wmma_bf16(a_cur, b_cur[j], acc[j]);
    a_cur = a_nxt;
#pragma unroll
    for (int j = 0; j < 4; ++j) b_cur[j] = b_nxt[j];
  }
#pragma unroll
  for (int j = 0; j < 4; ++j) acc[j] = wmma_bf16(a_cur, b_cur[j], acc[j]);

#pragma unroll
  for (int j = 0; j < 4; ++j) {
    const int n = nbase + j * 16 + r;
#pragma unroll
    for (int i = 0; i < 8; ++i) {
      const int m = mtile + i + hh * 8;       // C-layout row mapping
      float v = acc[j][i] + bv[j];
      if (MODE == 0) {
        outf[(size_t)m * DIMN + n] = v;
      } else {
        const int b = m >> 11, t = m & (TSEQ - 1);
        const int h = n >> 6,  d = n & (HD - 1);
        if (MODE == 1)
          outb[(((size_t)(b * NH + h) * TSEQ) + t) * HD + d] = f2bf(v);
        else
          outb[(((size_t)(b * NH + h) * HD) + d) * TSEQ + t] = f2bf(v);
      }
    }
  }
}

// ---------------------------------------------------------------------------
// Flash attention: one wave owns a 16-row T tile of one (b,h).
// Software-pipelined: next chunk's K operands + current chunk's V operands are
// issued before the score WMMAs; the softmax VALU block overlaps the loads.
__global__ __launch_bounds__(128, 1) void attn_kernel(
    const unsigned short* __restrict__ Qb, const unsigned short* __restrict__ Kb,
    const unsigned short* __restrict__ Vt, unsigned short* __restrict__ Ob) {
  __shared__ __attribute__((aligned(16))) unsigned short pl[4][16 * 48];

  const int lane = threadIdx.x & 31;
  const int wv   = threadIdx.x >> 5;
  const int r    = lane & 15;
  const int hh   = lane >> 4;
  const int bh   = blockIdx.y;            // 0..63
  const int b    = bh >> 4, h = bh & 15;
  const int t0   = blockIdx.x * 64 + wv * 16;

  const unsigned short* Qp = Qb + (size_t)bh * TSEQ * HD;
  const unsigned short* Kp = Kb + (size_t)bh * TSEQ * HD;
  const unsigned short* Vp = Vt + (size_t)bh * HD * TSEQ;
  unsigned short* lp = pl[wv];

  // Q operands for hd = [0,32) and [32,64), resident for the whole S loop.
  const size_t qrow = (size_t)(t0 + r) * HD;
  const bf16x16 qa0 = load2(Qp + qrow + 0  + hh * 16, Qp + qrow + 0  + hh * 16 + 8);
  const bf16x16 qa1 = load2(Qp + qrow + 32 + hh * 16, Qp + qrow + 32 + hh * 16 + 8);

  v8f acc[4] = {v8f{}, v8f{}, v8f{}, v8f{}};
  float rmax[8], rsum[8];
#pragma unroll
  for (int i = 0; i < 8; ++i) { rmax[i] = -1e30f; rsum[i] = 0.f; }

  const float scale = 0.125f;  // Hd^-0.5

  // K-operand loader for a 32-wide s-chunk (two 16-col score subtiles).
  auto loadK = [&](int s, bf16x16* kt) {
    const size_t k0row = (size_t)(s + r) * HD;
    const size_t k1row = (size_t)(s + 16 + r) * HD;
    kt[0] = load2(Kp + k0row +      hh * 16, Kp + k0row +      hh * 16 + 8);
    kt[1] = load2(Kp + k0row + 32 + hh * 16, Kp + k0row + 32 + hh * 16 + 8);
    kt[2] = load2(Kp + k1row +      hh * 16, Kp + k1row +      hh * 16 + 8);
    kt[3] = load2(Kp + k1row + 32 + hh * 16, Kp + k1row + 32 + hh * 16 + 8);
  };

  bf16x16 kc[4], kn[4];
  loadK(0, kc);

  for (int s0 = 0; s0 < TSEQ; s0 += 32) {
    // ---- issue current-chunk V and next-chunk K loads first (pipelining)
    bf16x16 vb[4];
#pragma unroll
    for (int j = 0; j < 4; ++j) {
      const size_t vrow = (size_t)(j * 16 + r) * TSEQ + s0 + hh * 16;
      vb[j] = load2(Vp + vrow, Vp + vrow + 8);
    }
    const int snxt = (s0 + 32 < TSEQ) ? s0 + 32 : 0;  // branch-free clamp
    loadK(snxt, kn);
    if (s0 + 64 < TSEQ) {     // global_prefetch_b8 two chunks ahead
      __builtin_prefetch(Kp + (size_t)(s0 + 64 + r) * HD, 0, 0);
      __builtin_prefetch(Vp + (size_t)r * TSEQ + s0 + 64, 0, 0);
    }

    // ---- scores: two 16-wide s-subtiles, K=64 -> 2 WMMAs each
    v8f c0{}, c1{};
    c0 = wmma_bf16(qa0, kc[0], c0); c0 = wmma_bf16(qa1, kc[1], c0);
    c1 = wmma_bf16(qa0, kc[2], c1); c1 = wmma_bf16(qa1, kc[3], c1);

    // ---- online softmax (row = i + hh*8; 16-lane groups hold one row)
#pragma unroll
    for (int i = 0; i < 8; ++i) {
      float s0v = c0[i] * scale, s1v = c1[i] * scale;
      float mx = fmaxf(s0v, s1v);
#pragma unroll
      for (int off = 1; off < 16; off <<= 1)
        mx = fmaxf(mx, __shfl_xor(mx, off, 32));
      float nm   = fmaxf(rmax[i], mx);
      float corr = __expf(rmax[i] - nm);
      rmax[i]    = nm;
      float p0 = __expf(s0v - nm), p1 = __expf(s1v - nm);
      float ps = p0 + p1;
#pragma unroll
      for (int off = 1; off < 16; off <<= 1)
        ps += __shfl_xor(ps, off, 32);
      rsum[i] = rsum[i] * corr + ps;
      acc[0][i] *= corr; acc[1][i] *= corr;
      acc[2][i] *= corr; acc[3][i] *= corr;
      const int rr = i + hh * 8;
      lp[rr * 48 + r]      = f2bf(p0);   // P tile, row-major bf16 in LDS
      lp[rr * 48 + 16 + r] = f2bf(p1);
    }
    __syncthreads();

    // ---- re-layout P(16x32) as A operand from LDS, then P @ V
    bf16x16 pa = load2(lp + r * 48 + hh * 16, lp + r * 48 + hh * 16 + 8);
#pragma unroll
    for (int j = 0; j < 4; ++j) acc[j] = wmma_bf16(pa, vb[j], acc[j]);
    __syncthreads();

#pragma unroll
    for (int j = 0; j < 4; ++j) kc[j] = kn[j];
  }

  // ---- normalize and scatter O as bf16 rows (b,t, h*64+d)
#pragma unroll
  for (int j = 0; j < 4; ++j) {
#pragma unroll
    for (int i = 0; i < 8; ++i) {
      const int rr = i + hh * 8;
      const float v = acc[j][i] / rsum[i];
      const int t = t0 + rr;
      Ob[((size_t)(b * TSEQ + t)) * DIMN + h * HD + j * 16 + r] = f2bf(v);
    }
  }
}

// ---------------------------------------------------------------------------
extern "C" void kernel_launch(void* const* d_in, const int* in_sizes, int n_in,
                              void* d_out, int out_size, void* d_ws, size_t ws_size,
                              hipStream_t stream) {
  const float* x   = (const float*)d_in[0];  // (4,2048,1024)
  const float* ctx = (const float*)d_in[1];  // (4,2048,768)
  const float* Wq  = (const float*)d_in[2];
  const float* bq  = (const float*)d_in[3];
  const float* Wk  = (const float*)d_in[4];
  const float* bk  = (const float*)d_in[5];
  const float* Wv  = (const float*)d_in[6];
  const float* bv  = (const float*)d_in[7];
  const float* Wo  = (const float*)d_in[8];
  const float* bo  = (const float*)d_in[9];
  float* out = (float*)d_out;

  const int M    = BSZ * TSEQ;               // 8192
  const int NX   = M * DIMN;                 // x / Q / O elements
  const int NC   = M * CTXD;                 // context elements

  unsigned char* ws = (unsigned char*)d_ws;
  size_t off = 0;
  auto alloc = [&](size_t bytes) {
    void* p = ws + off;
    off += (bytes + 255) & ~(size_t)255;
    return p;
  };
  unsigned short* xb  = (unsigned short*)alloc((size_t)NX * 2);  // reused as O
  unsigned short* cb  = (unsigned short*)alloc((size_t)NC * 2);
  unsigned short* Wqt = (unsigned short*)alloc((size_t)DIMN * DIMN * 2);
  unsigned short* Wkt = (unsigned short*)alloc((size_t)DIMN * CTXD * 2);
  unsigned short* Wvt = (unsigned short*)alloc((size_t)DIMN * CTXD * 2);
  unsigned short* Wot = (unsigned short*)alloc((size_t)DIMN * DIMN * 2);
  unsigned short* Qb  = (unsigned short*)alloc((size_t)NX * 2);
  unsigned short* Kb  = (unsigned short*)alloc((size_t)NX * 2);
  unsigned short* Vtb = (unsigned short*)alloc((size_t)NX * 2);
  unsigned short* Ob  = xb;   // xb dead after the Q projection

  // 1) converts
  cvt_bf16<<<NX / 256, 256, 0, stream>>>(x, xb, NX);
  cvt_bf16<<<NC / 256, 256, 0, stream>>>(ctx, cb, NC);
  cvt_tr<<<(DIMN * DIMN) / 256, 256, 0, stream>>>(Wq, Wqt, DIMN);
  cvt_tr<<<(DIMN * CTXD) / 256, 256, 0, stream>>>(Wk, Wkt, CTXD);
  cvt_tr<<<(DIMN * CTXD) / 256, 256, 0, stream>>>(Wv, Wvt, CTXD);
  cvt_tr<<<(DIMN * DIMN) / 256, 256, 0, stream>>>(Wo, Wot, DIMN);

  // 2) projections (WMMA)
  dim3 ggrid(M / 16, DIMN / 512);
  gemm_bf16<1><<<ggrid, 256, 0, stream>>>(xb, Wqt, bq, Qb,  nullptr, M, DIMN);
  gemm_bf16<1><<<ggrid, 256, 0, stream>>>(cb, Wkt, bk, Kb,  nullptr, M, CTXD);
  gemm_bf16<2><<<ggrid, 256, 0, stream>>>(cb, Wvt, bv, Vtb, nullptr, M, CTXD);

  // 3) attention (WMMA flash, pipelined)
  attn_kernel<<<dim3(TSEQ / 64, BSZ * NH), 128, 0, stream>>>(Qb, Kb, Vtb, Ob);

  // 4) output projection (WMMA, f32 out)
  gemm_bf16<0><<<ggrid, 256, 0, stream>>>(Ob, Wot, bo, nullptr, out, M, DIMN);
}
